// gatedGNN_86500641341508
// MI455X (gfx1250) — matrix operand order repaired and checked by reading, not compiled, not run on hardware
//
#include <hip/hip_runtime.h>
#include <hip/hip_bf16.h>
#include <stdint.h>

#define N_NODES 100000
#define NEDGE   3200000
#define NFEAT   256
#define NHID    256
#define NOUT    128

// async global->LDS staging of the B tile (CDNA5 GLOBAL_LOAD_ASYNC_TO_LDS_B128).
#define GG_USE_ASYNC 1

typedef __attribute__((ext_vector_type(16))) __bf16 v16bf;
typedef __attribute__((ext_vector_type(8)))  __bf16 v8bf;
typedef __attribute__((ext_vector_type(8)))  float  v8f;

__device__ __forceinline__ unsigned short f2bf(float f) {
  union { float f; unsigned int u; } c; c.f = f;
  unsigned int u = c.u;
  unsigned int r = u + 0x7FFFu + ((u >> 16) & 1u);   // round-to-nearest-even
  return (unsigned short)(r >> 16);
}

// ---------------------------------------------------------------- zero fill
__global__ void gg_zero_f32(float* __restrict__ p, int n) {
  for (int i = blockIdx.x * blockDim.x + threadIdx.x; i < n;
       i += gridDim.x * blockDim.x)
    p[i] = 0.0f;
}

// ------------------------------------------------------- f32 -> bf16 convert
__global__ void gg_cvt_bf16(const float* __restrict__ in,
                            unsigned short* __restrict__ out, int n) {
  for (int i = blockIdx.x * blockDim.x + threadIdx.x; i < n;
       i += gridDim.x * blockDim.x)
    out[i] = f2bf(in[i]);
}

// ------------------------------------------- pack [Wself|Wnb] into B-fragments
// layout: wp[layer*131072 + (((ntb*8 + kt)*8 + ntin)*512) + lane*16 + i]
//   = W[kt*32 + (lane/16)*16 + i][(ntb*8+ntin)*16 + lane%16]   (bf16)
// ntb in 0..3 groups 128 output columns -> each block's B tile is one
// contiguous 64KB region (8 kt x 8 ntin x 1KB).
__global__ void gg_pack_w(const float* __restrict__ Ws1, const float* __restrict__ Wn1,
                          const float* __restrict__ Ws2, const float* __restrict__ Wn2,
                          unsigned short* __restrict__ wp) {
  int tid = blockIdx.x * blockDim.x + threadIdx.x;
  if (tid >= 2 * 131072) return;
  int layer = tid >> 17;
  int t     = tid & 131071;
  int frag  = t >> 9;                 // 0..255
  int li    = t & 511;
  int lane  = li >> 4, i = li & 15;
  int ntb  = frag >> 6;               // 0..3
  int kt   = (frag >> 3) & 7;         // 0..7
  int ntin = frag & 7;                // 0..7
  int nt   = ntb * 8 + ntin;
  int k    = kt * 32 + (lane >> 4) * 16 + i;
  int col  = nt * 16 + (lane & 15);
  const float* Ws = layer ? Ws2 : Ws1;
  const float* Wn = layer ? Wn2 : Wn1;
  float v = (col < 256) ? Ws[k * 256 + col] : Wn[k * 256 + (col - 256)];
  wp[tid] = f2bf(v);
}

// --------------------------------------------------------------- dual GEMM
// C[N,512] = A[N,256](bf16) @ Bpacked(256x512 bf16), split into C0 (cols 0-255)
// and C1 (cols 256-511). Wave tile = 16 rows x 128 cols (8 accumulators),
// block = 8 waves = 128 rows x 128 cols. B tile (64KB) staged in LDS once per
// block via async global->LDS copies. Per kt-step: issue all 16 ds_loads as a
// batch, then run the 8 independent WMMAs back-to-back.
__global__ __launch_bounds__(256) void gg_gemm_dual(
    const unsigned short* __restrict__ Abf,
    const unsigned short* __restrict__ Bp,
    float* __restrict__ C0, float* __restrict__ C1, int nrows) {
  __shared__ unsigned short sB[32768];                 // 64KB
  const int tid  = threadIdx.x;
  const int lane = tid & 31;
  const int wave = tid >> 5;
  const int rowBase = blockIdx.x * 128 + wave * 16;
  const int colBase = blockIdx.y * 128;                // 0,128,256,384
  float* __restrict__ C = (colBase < 256) ? C0 : C1;
  const int cb = colBase & 255;

  // ---- stage this block's 64KB B tile into LDS
  {
    const unsigned short* bsrc = Bp + (size_t)blockIdx.y * 32768;
#if GG_USE_ASYNC
    unsigned ldsbase = (unsigned)(uintptr_t)(&sB[0]) + (unsigned)tid * 256u;
    unsigned long long ga =
        (unsigned long long)(uintptr_t)(bsrc + (size_t)tid * 128);
#pragma unroll
    for (int j = 0; j < 16; ++j) {
      // INST_OFFSET applies to both LDS dest and global src (ISA 08 §4.4)
      asm volatile("global_load_async_to_lds_b128 %0, %1, off offset:%c2"
                   :: "v"(ldsbase), "v"(ga), "i"(j * 16) : "memory");
    }
    asm volatile("s_wait_asynccnt 0x0" ::: "memory");
#else
    const float4* s4 = (const float4*)bsrc;
    float4* d4 = (float4*)sB;
#pragma unroll
    for (int j = 0; j < 16; ++j)
      d4[tid * 16 + j] = s4[tid * 16 + j];
#endif
  }
  __syncthreads();

  v8f acc[8] = {};

  int arow = rowBase + (lane & 15);
  if (arow >= nrows) arow = nrows - 1;                 // clamp; store is guarded
  const int khalf = (lane >> 4) ? 16 : 0;
  const unsigned short* arowp = Abf + (size_t)arow * 256 + khalf;

#pragma unroll
  for (int kt = 0; kt < 8; ++kt) {
    const unsigned short* aptr = arowp + kt * 32;
    v8bf alo = *(const v8bf*)(aptr);                   // K = k0 .. k0+7
    v8bf ahi = *(const v8bf*)(aptr + 8);               // K = k0+8 .. k0+15
    v16bf a = __builtin_shufflevector(alo, ahi, 0, 1, 2, 3, 4, 5, 6, 7,
                                      8, 9, 10, 11, 12, 13, 14, 15);
    const unsigned short* bb = &sB[(kt * 8) * 512 + lane * 16];
    // batch all B-fragment loads first (independent ds_loads -> one wait)
    v16bf bfr[8];
#pragma unroll
    for (int t = 0; t < 8; ++t)
      bfr[t] = *(const v16bf*)(bb + t * 512);
    // then the 8 independent WMMAs back-to-back
#pragma unroll
    for (int t = 0; t < 8; ++t)
      acc[t] = __builtin_amdgcn_wmma_f32_16x16x32_bf16(
          false, a, false, bfr[t], (short)0, acc[t], false, false);
  }

  // C/D layout: VGPR i, lanes 0-15 -> M=i, lanes 16-31 -> M=8+i, N=lane%16
  const int mOff = (lane >> 4) * 8;
  const int nCol = lane & 15;
#pragma unroll
  for (int t = 0; t < 8; ++t) {
#pragma unroll
    for (int i = 0; i < 8; ++i) {
      int r = rowBase + mOff + i;
      if (r < nrows)
        C[(size_t)r * 256 + cb + t * 16 + nCol] = acc[t][i];
    }
  }
}

// ------------------------------------------------------------- edge scatter
// actnb[row] += val * support[col]; 64 lanes (float4 each) per edge.
// support table (102MB) is L2-resident; atomics resolve in L2.
__global__ void gg_scatter(const int* __restrict__ erow, const int* __restrict__ ecol,
                           const float* __restrict__ eval,
                           const float* __restrict__ support,
                           float* __restrict__ actnb) {
  long long gid = (long long)blockIdx.x * blockDim.x + threadIdx.x;
  int e = (int)(gid >> 6);
  int q = (int)(gid & 63);
  if (e >= NEDGE) return;
  int row = erow[e], col = ecol[e];
  float val = eval[e];
  const float4 s = *(const float4*)(support + (size_t)col * 256 + q * 4);
  float* dst = actnb + (size_t)row * 256 + q * 4;
  atomicAdd(dst + 0, val * s.x);
  atomicAdd(dst + 1, val * s.y);
  atomicAdd(dst + 2, val * s.z);
  atomicAdd(dst + 3, val * s.w);
}

// ------------------------------------------------------- bias + relu (f4)
__global__ void gg_bias_relu(const float* __restrict__ a, const float* __restrict__ b,
                             const float* __restrict__ bias, float* __restrict__ out,
                             int n4) {
  int i = blockIdx.x * blockDim.x + threadIdx.x;
  if (i >= n4) return;
  int base = i * 4;
  float4 va = *(const float4*)(a + base);
  float4 vb = *(const float4*)(b + base);
  int f = base & 255;
  float4 r;
  r.x = fmaxf(va.x + vb.x + bias[f + 0], 0.f);
  r.y = fmaxf(va.y + vb.y + bias[f + 1], 0.f);
  r.z = fmaxf(va.z + vb.z + bias[f + 2], 0.f);
  r.w = fmaxf(va.w + vb.w + bias[f + 3], 0.f);
  *(float4*)(out + base) = r;
}

// --------------------------------------------------- gate logits (wave/node)
__global__ void gg_gate_dots(const float* __restrict__ x,
                             const float* __restrict__ g1w, const float* __restrict__ g1b,
                             const float* __restrict__ g2w, const float* __restrict__ g2b,
                             float* __restrict__ g1, float* __restrict__ g2) {
  int wid = (blockIdx.x * blockDim.x + threadIdx.x) >> 5;
  int lane = threadIdx.x & 31;
  if (wid >= N_NODES) return;
  const float* xr = x + (size_t)wid * 256;
  float s1 = 0.f, s2 = 0.f;
#pragma unroll
  for (int k = lane; k < 256; k += 32) {
    float xv = xr[k];
    s1 += xv * g1w[k];
    s2 += xv * g2w[k];
  }
#pragma unroll
  for (int off = 16; off; off >>= 1) {
    s1 += __shfl_down(s1, off, 32);
    s2 += __shfl_down(s2, off, 32);
  }
  if (lane == 0) { g1[wid] = s1 + g1b[0]; g2[wid] = s2 + g2b[0]; }
}

// ------------------------------------------------------ softmax reductions
__global__ void gg_rmax_part(const float* __restrict__ g1, const float* __restrict__ g2,
                             float* __restrict__ p1, float* __restrict__ p2, int n) {
  __shared__ float s1[256], s2[256];
  float m1 = -3.4e38f, m2 = -3.4e38f;
  for (int i = blockIdx.x * blockDim.x + threadIdx.x; i < n;
       i += gridDim.x * blockDim.x) {
    m1 = fmaxf(m1, g1[i]); m2 = fmaxf(m2, g2[i]);
  }
  int t = threadIdx.x;
  s1[t] = m1; s2[t] = m2; __syncthreads();
  for (int o = 128; o; o >>= 1) {
    if (t < o) { s1[t] = fmaxf(s1[t], s1[t + o]); s2[t] = fmaxf(s2[t], s2[t + o]); }
    __syncthreads();
  }
  if (t == 0) { p1[blockIdx.x] = s1[0]; p2[blockIdx.x] = s2[0]; }
}

__global__ void gg_rmax_final(const float* __restrict__ p1, const float* __restrict__ p2,
                              float* __restrict__ scal, int n) {
  __shared__ float s1[256], s2[256];
  int t = threadIdx.x;
  s1[t] = (t < n) ? p1[t] : -3.4e38f;
  s2[t] = (t < n) ? p2[t] : -3.4e38f;
  __syncthreads();
  for (int o = 128; o; o >>= 1) {
    if (t < o) { s1[t] = fmaxf(s1[t], s1[t + o]); s2[t] = fmaxf(s2[t], s2[t + o]); }
    __syncthreads();
  }
  if (t == 0) { scal[0] = s1[0]; scal[1] = s2[0]; }
}

__global__ void gg_exp_part(float* __restrict__ g1, float* __restrict__ g2,
                            const float* __restrict__ scal,
                            float* __restrict__ p1, float* __restrict__ p2, int n) {
  __shared__ float s1[256], s2[256];
  float m1 = scal[0], m2 = scal[1];
  float a1 = 0.f, a2 = 0.f;
  for (int i = blockIdx.x * blockDim.x + threadIdx.x; i < n;
       i += gridDim.x * blockDim.x) {
    float e1 = __expf(g1[i] - m1); g1[i] = e1; a1 += e1;
    float e2 = __expf(g2[i] - m2); g2[i] = e2; a2 += e2;
  }
  int t = threadIdx.x;
  s1[t] = a1; s2[t] = a2; __syncthreads();
  for (int o = 128; o; o >>= 1) {
    if (t < o) { s1[t] += s1[t + o]; s2[t] += s2[t + o]; }
    __syncthreads();
  }
  if (t == 0) { p1[blockIdx.x] = s1[0]; p2[blockIdx.x] = s2[0]; }
}

__global__ void gg_rsum_final(const float* __restrict__ p1, const float* __restrict__ p2,
                              float* __restrict__ scal, int n) {
  __shared__ float s1[256], s2[256];
  int t = threadIdx.x;
  s1[t] = (t < n) ? p1[t] : 0.f;
  s2[t] = (t < n) ? p2[t] : 0.f;
  __syncthreads();
  for (int o = 128; o; o >>= 1) {
    if (t < o) { s1[t] += s1[t + o]; s2[t] += s2[t + o]; }
    __syncthreads();
  }
  if (t == 0) { scal[2] = s1[0]; scal[3] = s2[0]; }
}

// ------------------------------------------- xr = x1 * r (bf16 for GEMM2)
__global__ void gg_make_xr(const float* __restrict__ x1, const float* __restrict__ g1,
                           const float* __restrict__ scal,
                           unsigned short* __restrict__ xrbf, int n) {
  float inv = 1.0f / scal[2];
  for (int i = blockIdx.x * blockDim.x + threadIdx.x; i < n;
       i += gridDim.x * blockDim.x) {
    int row = i >> 8;
    xrbf[i] = f2bf(x1[i] * (g1[row] * inv));
  }
}

// ------------------------- x_enc = (1-z)*x1 + z*relu(aself2+anb2+b2); colsum
__global__ void gg_combine(const float* __restrict__ x1, const float* __restrict__ aself2,
                           const float* __restrict__ anb2, const float* __restrict__ b2,
                           const float* __restrict__ g2, const float* __restrict__ scal,
                           float* __restrict__ xenc, float* __restrict__ colsum) {
  int f = threadIdx.x;
  int row0 = blockIdx.x * 64;
  float inv = 1.0f / scal[3];
  float bf = b2[f];
  float acc = 0.f;
  for (int r = 0; r < 64; ++r) {
    int row = row0 + r;
    if (row < N_NODES) {
      float z = g2[row] * inv;
      size_t idx = (size_t)row * 256 + f;
      float x2 = fmaxf(aself2[idx] + anb2[idx] + bf, 0.f);
      float xe = (1.f - z) * x1[idx] + z * x2;
      xenc[idx] = xe;
      acc += xe;
    }
  }
  atomicAdd(colsum + f, acc);
}

// ----------------------------------------- out = mean(x_enc) @ e2p_w + e2p_b
__global__ void gg_final_out(const float* __restrict__ colsum,
                             const float* __restrict__ e2pw, const float* __restrict__ e2pb,
                             float* __restrict__ out) {
  int j = threadIdx.x;
  if (j >= NOUT) return;
  float s = e2pb[j];
  const float invN = 1.0f / (float)N_NODES;
#pragma unroll 8
  for (int k = 0; k < 256; ++k)
    s += (colsum[k] * invN) * e2pw[k * NOUT + j];
  out[j] = s;
}

extern "C" void kernel_launch(void* const* d_in, const int* in_sizes, int n_in,
                              void* d_out, int out_size, void* d_ws, size_t ws_size,
                              hipStream_t stream) {
  const float* x      = (const float*)d_in[0];
  const int*   erow   = (const int*)d_in[1];
  const int*   ecol   = (const int*)d_in[2];
  const float* eval   = (const float*)d_in[3];
  const float* Wself1 = (const float*)d_in[4];
  const float* Wnb1   = (const float*)d_in[5];
  const float* b1     = (const float*)d_in[6];
  const float* Wself2 = (const float*)d_in[7];
  const float* Wnb2   = (const float*)d_in[8];
  const float* b2     = (const float*)d_in[9];
  const float* g1w    = (const float*)d_in[10];
  const float* g1b    = (const float*)d_in[11];
  const float* g2w    = (const float*)d_in[12];
  const float* g2b    = (const float*)d_in[13];
  const float* e2pw   = (const float*)d_in[14];
  const float* e2pb   = (const float*)d_in[15];
  float* out = (float*)d_out;

  const size_t NX = (size_t)N_NODES * 256;

  // ---- workspace carve-up
  char* ws = (char*)d_ws;
  size_t o = 0;
  auto carve = [&](size_t bytes) { size_t r = o; o = (o + bytes + 255) & ~(size_t)255; return r; };
  size_t o_xbf = carve(NX * 2);                 // bf16 A (x, then reused for x*r)
  size_t o_wp  = carve(2 * 131072 * 2);         // packed weights, both layers
  size_t o_B1  = carve(NX * 4);                 // actself1 -> x1
  size_t o_B2  = carve(NX * 4);                 // support1 -> actself2
  size_t o_B3  = carve(NX * 4);                 // actnb1   -> support2
  size_t o_B4  = carve(NX * 4);                 // actnb2
  size_t o_g1  = carve((size_t)N_NODES * 4);
  size_t o_g2  = carve((size_t)N_NODES * 4);
  size_t o_pt  = carve(4 * 256 * 4);            // reduction partials
  size_t o_sc  = carve(4 * 4);                  // max1,max2,sum1,sum2
  size_t o_cs  = carve(256 * 4);                // column sums
  (void)ws_size;

  unsigned short* xbf = (unsigned short*)(ws + o_xbf);
  unsigned short* wp  = (unsigned short*)(ws + o_wp);
  float* B1 = (float*)(ws + o_B1);
  float* B2 = (float*)(ws + o_B2);
  float* B3 = (float*)(ws + o_B3);
  float* B4 = (float*)(ws + o_B4);
  float* g1 = (float*)(ws + o_g1);
  float* g2 = (float*)(ws + o_g2);
  float* pm1 = (float*)(ws + o_pt);
  float* pm2 = pm1 + 256;
  float* ps1 = pm1 + 512;
  float* ps2 = pm1 + 768;
  float* scal = (float*)(ws + o_sc);
  float* colsum = (float*)(ws + o_cs);

  const int T = 256;
  const int nx_blocks = (int)((NX + T - 1) / T);

  // 0) zero the atomic-destination buffers + colsum
  gg_zero_f32<<<2048, T, 0, stream>>>(B3, (int)NX);
  gg_zero_f32<<<2048, T, 0, stream>>>(B4, (int)NX);
  gg_zero_f32<<<1, T, 0, stream>>>(colsum, 256);

  // 1) convert x to bf16; pack both layers' weights into WMMA-B fragments
  gg_cvt_bf16<<<nx_blocks, T, 0, stream>>>(x, xbf, (int)NX);
  gg_pack_w<<<(2 * 131072 + T - 1) / T, T, 0, stream>>>(Wself1, Wnb1, Wself2, Wnb2, wp);

  // 2) GEMM1: [actself1 | support1] = x @ [Wself1 | Wnb1]
  dim3 ggrid((N_NODES + 127) / 128, 4);
  gg_gemm_dual<<<ggrid, T, 0, stream>>>(xbf, wp, B1, B2, N_NODES);

  // 3) edge scatter 1: actnb1 += val * support1[col]
  long long sthreads = (long long)NEDGE * 64;
  gg_scatter<<<(unsigned)((sthreads + T - 1) / T), T, 0, stream>>>(erow, ecol, eval, B2, B3);

  // 4) x1 = relu(actself1 + actnb1 + b1)   (in place over B1)
  gg_bias_relu<<<(int)(NX / 4 + T - 1) / T, T, 0, stream>>>(B1, B3, b1, B1, (int)(NX / 4));

  // 5) gate logits, softmax over nodes
  gg_gate_dots<<<(N_NODES * 32 + T - 1) / T, T, 0, stream>>>(B1, g1w, g1b, g2w, g2b, g1, g2);
  gg_rmax_part<<<256, T, 0, stream>>>(g1, g2, pm1, pm2, N_NODES);
  gg_rmax_final<<<1, T, 0, stream>>>(pm1, pm2, scal, 256);
  gg_exp_part<<<256, T, 0, stream>>>(g1, g2, scal, ps1, ps2, N_NODES);
  gg_rsum_final<<<1, T, 0, stream>>>(ps1, ps2, scal, 256);

  // 6) xr = bf16(x1 * r)   (reuse xbf)
  gg_make_xr<<<nx_blocks, T, 0, stream>>>(B1, g1, scal, xbf, (int)NX);

  // 7) GEMM2: [actself2 | support2] = xr @ [Wself2 | Wnb2]
  gg_gemm_dual<<<ggrid, T, 0, stream>>>(xbf, wp + 131072, B2, B3, N_NODES);

  // 8) edge scatter 2: actnb2 += val * support2[col]
  gg_scatter<<<(unsigned)((sthreads + T - 1) / T), T, 0, stream>>>(erow, ecol, eval, B3, B4);

  // 9) combine -> x_enc (straight into d_out) + column sums
  gg_combine<<<(N_NODES + 63) / 64, T, 0, stream>>>(B1, B2, B4, b2, g2, scal, out, colsum);

  // 10) out row: mean(x_enc) @ e2p_w + e2p_b
  gg_final_out<<<1, 128, 0, stream>>>(colsum, e2pw, e2pb, out + NX);
}
